// BiLSTM_CRFModel_31842887533087
// MI455X (gfx1250) — compile-verified, matrix operand order checked by reading, not run on hardware
//
#include <hip/hip_runtime.h>
#include <hip/hip_bf16.h>
#include <math.h>

typedef __attribute__((ext_vector_type(16))) __bf16 v16bf;
typedef __attribute__((ext_vector_type(8)))  float  v8f;
typedef __attribute__((ext_vector_type(4)))  unsigned int u32x4;
typedef __attribute__((ext_vector_type(8)))  unsigned int u32x8;

constexpr int B  = 64;
constexpr int T  = 512;
constexpr int D  = 300;
constexpr int Dp = 320;   // D padded to multiple of 32 (WMMA K)
constexpr int H  = 256;
constexpr int NH = 1024;  // 4*H
constexpr int C  = 9;
constexpr int BT = B * T; // 32768

constexpr unsigned LSTM_SMEM = 32768 + 65536 + 65536 + 256; // h_bf16 + h_f32 + c_f32 + len

__device__ __forceinline__ unsigned short f2bf(float x) {
    unsigned int u = __float_as_uint(x);
    return (unsigned short)((u + 0x7FFFu + ((u >> 16) & 1u)) >> 16);
}
__device__ __forceinline__ float bf2f(unsigned short h) {
    return __uint_as_float(((unsigned int)h) << 16);
}

union FragH { v16bf v; unsigned int u[8]; };
union FragF { v8f  v; float        f[8]; };

// Load one 16x32 bf16 fragment (A-layout; B identical when W stored [N][K]).
// ISA layout: lanes 0-15 -> rows 0-15, K 0..7 & 16..23; lanes 16-31 -> K 8..15 & 24..31.
__device__ __forceinline__ FragH load_frag(const unsigned short* p, int ld) {
    int lane = threadIdx.x & 31;
    int r = lane & 15, sel = lane >> 4;
    const unsigned short* row = p + (size_t)r * ld + sel * 8;
    FragH f;
#pragma unroll
    for (int j = 0; j < 4; ++j)
        f.u[j] = *(const unsigned int*)(row + 2 * j);
#pragma unroll
    for (int j = 0; j < 4; ++j)
        f.u[4 + j] = *(const unsigned int*)(row + 16 + 2 * j);
    return f;
}

// Issue one Tensor Data Mover 2D tile load (bf16 elements) into LDS.
// Wave-uniform; D# groups 0/1 built in SGPRs (VADDR2/3 = NULL -> 2D form).
__device__ __forceinline__ void tdm_load_2d(unsigned lds_off, const void* gptr,
                                            unsigned tile_x, unsigned tile_y,
                                            unsigned stride_elts) {
    unsigned long long ga = (unsigned long long)gptr;
    unsigned dim1 = 1u << 20;  // generous bounds; tiles are always in-range
    u32x4 g0;
    g0[0] = 1u;                                               // count=1, user descriptor
    g0[1] = lds_off;                                          // LDS byte address
    g0[2] = (unsigned)ga;                                     // global_addr[31:0]
    g0[3] = (unsigned)((ga >> 32) & 0x01FFFFFFu) | (2u << 30);// global_addr[56:32], type=2
    u32x8 g1;
    g1[0] = 0x00010000u;                                      // data_size=1 (2 bytes)
    g1[1] = (stride_elts & 0xFFFFu) << 16;                    // tensor_dim0 lo16
    g1[2] = ((stride_elts >> 16) & 0xFFFFu) | ((dim1 & 0xFFFFu) << 16); // dim0 hi / dim1 lo
    g1[3] = ((dim1 >> 16) & 0xFFFFu) | (tile_x << 16);        // dim1 hi, tile_dim0
    g1[4] = tile_y & 0xFFFFu;                                 // tile_dim1 (tile_dim2=0)
    g1[5] = stride_elts;                                      // tensor_dim0_stride lo32
    g1[6] = 0u;                                               // stride0 hi, stride1 lo
    g1[7] = 0u;                                               // stride1 hi
    asm volatile("tensor_load_to_lds %0, %1" :: "s"(g0), "s"(g1) : "memory");
}

// ---------- weight convert: f32 [K x N] row-major -> bf16 [N x Kp] (transposed, K zero-padded)
__global__ void k_convT(const float* __restrict__ src, unsigned short* __restrict__ dst,
                        int K, int N, int Kp) {
    int i = blockIdx.x * blockDim.x + threadIdx.x;
    if (i >= N * Kp) return;
    int n = i / Kp, k = i % Kp;
    float v = (k < K) ? src[(size_t)k * N + n] : 0.f;
    dst[(size_t)n * Kp + k] = f2bf(v);
}

// ---------- embedding gather -> bf16 [BT x Dp], cols D..Dp-1 zero
__global__ void k_embed(const int* __restrict__ tok, const float* __restrict__ emb,
                        unsigned short* __restrict__ out) {
    int i = blockIdx.x * blockDim.x + threadIdx.x;
    if (i >= BT * Dp) return;
    int bt = i / Dp, d = i % Dp;
    float v = (d < D) ? emb[(size_t)tok[bt] * D + d] : 0.f;
    out[(size_t)bt * Dp + d] = f2bf(v);
}

// ---------- xz = emb @ Wk + b   (A: [BT x Dp] bf16, Wt: [NH x Dp] bf16, out: [BT x NH] f32)
// block = 256 thr (8 waves), tile 128x128, K-chunks of 32.
// A tile staged by TDM (tensor_load_to_lds), B tile by async global->LDS copies.
__global__ void k_xz(const unsigned short* __restrict__ A, const unsigned short* __restrict__ Wt,
                     const float* __restrict__ bias, float* __restrict__ out) {
    __shared__ __align__(16) unsigned short aT[128 * 32];
    __shared__ __align__(16) unsigned short bT[128 * 32];
    int tid = threadIdx.x;
    int m0 = blockIdx.y * 128;
    int n0 = blockIdx.x * 128;
    int lane = tid & 31, w = tid >> 5;
    int wm = w & 3, wn = w >> 2;              // wave tile: 32(M) x 64(N)
    FragF acc[2][4];
#pragma unroll
    for (int i = 0; i < 2; ++i)
#pragma unroll
        for (int j = 0; j < 4; ++j)
#pragma unroll
            for (int r = 0; r < 8; ++r) acc[i][j].f[r] = 0.f;

    int row = tid >> 1, seg = (tid & 1) * 16;     // 16 bf16 = 32B per thread (B tile)
    unsigned ldsA = (unsigned)(unsigned long long)(void*)aT;
    unsigned ldsB = (unsigned)(unsigned long long)(void*)(bT + row * 32 + seg);

    for (int k0 = 0; k0 < Dp; k0 += 32) {
        // A tile: one TDM descriptor, issued by wave 0 only (EXEC-independent DMA)
        if (w == 0)
            tdm_load_2d(ldsA, A + (size_t)m0 * Dp + k0, 32u, 128u, (unsigned)Dp);
        // B tile: per-lane async copies, 32B per thread (same IOFFSET applies to LDS+global)
        {
            unsigned voff = (unsigned)(((size_t)(n0 + row) * Dp + k0 + seg) * 2u);
            asm volatile(
                "global_load_async_to_lds_b128 %0, %1, %2 offset:0\n\t"
                "global_load_async_to_lds_b128 %0, %1, %2 offset:16"
                :: "v"(ldsB), "v"(voff), "s"(Wt) : "memory");
        }
        asm volatile("s_wait_asynccnt 0x0" ::: "memory");
        if (w == 0) __builtin_amdgcn_s_wait_tensorcnt(0);
        __syncthreads();

        FragH a0 = load_frag(aT + (wm * 32 + 0) * 32, 32);
        FragH a1 = load_frag(aT + (wm * 32 + 16) * 32, 32);
#pragma unroll
        for (int j = 0; j < 4; ++j) {
            FragH bj = load_frag(bT + (wn * 64 + j * 16) * 32, 32);
            acc[0][j].v = __builtin_amdgcn_wmma_f32_16x16x32_bf16(
                false, a0.v, false, bj.v, (short)0, acc[0][j].v, false, false);
            acc[1][j].v = __builtin_amdgcn_wmma_f32_16x16x32_bf16(
                false, a1.v, false, bj.v, (short)0, acc[1][j].v, false, false);
        }
        __syncthreads();
    }
    int rsel = lane >> 4, nc = lane & 15;
#pragma unroll
    for (int j = 0; j < 4; ++j) {
        int gn = n0 + wn * 64 + j * 16 + nc;
        float bv = bias[gn];
#pragma unroll
        for (int i = 0; i < 2; ++i)
#pragma unroll
            for (int r = 0; r < 8; ++r) {
                int gm = m0 + wm * 32 + i * 16 + r + rsel * 8;
                out[(size_t)gm * NH + gn] = acc[i][j].f[r] + bv;
            }
    }
}

// ---------- persistent LSTM: grid.x = direction, 1024 threads (32 waves)
// wave w: M-half mh=w&1 (32 batch rows), H-block hb=w>>1 (16 h-cols, all 4 gates)
// -> each B fragment reused across 2 M-tiles; 32 B-frag loads + 64 WMMA per wave per step.
__global__ void k_lstm(const float* __restrict__ xzf, const float* __restrict__ xzb,
                       const unsigned short* __restrict__ WrTf, const unsigned short* __restrict__ WrTb,
                       const int* __restrict__ len,
                       unsigned short* __restrict__ houtf, unsigned short* __restrict__ houtb) {
    extern __shared__ unsigned char smem[];
    unsigned short* hbf = (unsigned short*)smem;                 // [B][H] bf16
    float* hf  = (float*)(smem + 32768);                         // [B][H] f32
    float* cf  = (float*)(smem + 32768 + 65536);                 // [B][H] f32
    int*   lenS= (int*)  (smem + 32768 + 65536 + 65536);         // [B]
    int dir = blockIdx.x;
    const float* xz = dir ? xzb : xzf;
    const unsigned short* WrT = dir ? WrTb : WrTf;
    unsigned short* hout = dir ? houtb : houtf;

    int tid = threadIdx.x, lane = tid & 31, w = tid >> 5;
    int mh = w & 1;          // batch rows [32*mh, 32*mh+32)
    int hb = w >> 1;         // h-cols [16*hb, 16*hb+16)
    int rsel = lane >> 4, nc = lane & 15;

    for (int i = tid; i < B * H; i += blockDim.x) { hbf[i] = 0; hf[i] = 0.f; cf[i] = 0.f; }
    for (int i = tid; i < B; i += blockDim.x) lenS[i] = len[i];
    __syncthreads();

    for (int t = 0; t < T; ++t) {
        int tt = dir ? (T - 1 - t) : t;
        // accumulators init = xz[:, tt, :]
        FragF acc[4][2];
#pragma unroll
        for (int g = 0; g < 4; ++g)
#pragma unroll
            for (int i = 0; i < 2; ++i) {
                int n = g * H + hb * 16 + nc;
#pragma unroll
                for (int r = 0; r < 8; ++r) {
                    int bm = mh * 32 + i * 16 + r + rsel * 8;
                    acc[g][i].f[r] = xz[((size_t)bm * T + tt) * NH + n];
                }
            }
        // h @ Wr : K=256 in 8 chunks; B frag shared across both M-tiles
#pragma unroll
        for (int kc = 0; kc < 8; ++kc) {
            FragH a0 = load_frag(hbf + (mh * 32 + 0) * H + kc * 32, H);
            FragH a1 = load_frag(hbf + (mh * 32 + 16) * H + kc * 32, H);
#pragma unroll
            for (int g = 0; g < 4; ++g) {
                FragH bfr = load_frag(WrT + (size_t)(g * H + hb * 16) * H + kc * 32, H);
                acc[g][0].v = __builtin_amdgcn_wmma_f32_16x16x32_bf16(
                    false, a0.v, false, bfr.v, (short)0, acc[g][0].v, false, false);
                acc[g][1].v = __builtin_amdgcn_wmma_f32_16x16x32_bf16(
                    false, a1.v, false, bfr.v, (short)0, acc[g][1].v, false, false);
            }
        }
        if (t + 1 < T) {  // hint next step's xz rows toward the caches
            int ttn = dir ? (T - 2 - t) : (t + 1);
            __builtin_prefetch(&xz[((size_t)(mh * 32 + rsel * 8) * T + ttn) * NH], 0, 3);
        }
        __syncthreads();  // all h reads done before anyone writes new h
        // gates + state update (each (bm, hcol) owned by exactly one lane)
        int hcol = hb * 16 + nc;
#pragma unroll
        for (int i = 0; i < 2; ++i) {
#pragma unroll
            for (int r = 0; r < 8; ++r) {
                int bm = mh * 32 + i * 16 + r + rsel * 8;
                float zi = acc[0][i].f[r], zf = acc[1][i].f[r];
                float zg = acc[2][i].f[r], zo = acc[3][i].f[r];
                float ig = 1.f / (1.f + __expf(-zi));
                float fg = 1.f / (1.f + __expf(-zf));
                float gg = tanhf(zg);
                float og = 1.f / (1.f + __expf(-zo));
                float cold = cf[bm * H + hcol];
                float cnew = fg * cold + ig * gg;
                float hnew = og * tanhf(cnew);
                bool msk = tt < lenS[bm];
                float cw = msk ? cnew : cold;
                float hw = msk ? hnew : hf[bm * H + hcol];
                cf[bm * H + hcol] = cw;
                hf[bm * H + hcol] = hw;
                unsigned short hbv = f2bf(hw);
                hbf[bm * H + hcol] = hbv;
                hout[((size_t)bm * T + tt) * H + hcol] = hbv;
            }
        }
        __syncthreads();  // writes visible before next iteration's reads
    }
}

// ---------- dense: logits[bt,c] = [h_f|h_b] @ dense_W + b   (C=9, VALU is plenty)
__global__ void k_dense(const unsigned short* __restrict__ hfm, const unsigned short* __restrict__ hbm,
                        const float* __restrict__ W, const float* __restrict__ bias,
                        float* __restrict__ out) {
    int i = blockIdx.x * blockDim.x + threadIdx.x;
    if (i >= BT * C) return;
    int bt = i / C, c = i % C;
    const unsigned short* hr = hfm + (size_t)bt * H;
    const unsigned short* br = hbm + (size_t)bt * H;
    float s = bias[c];
    for (int k = 0; k < H; ++k) s += bf2f(hr[k]) * W[(size_t)k * C + c];
    for (int k = 0; k < H; ++k) s += bf2f(br[k]) * W[(size_t)(H + k) * C + c];
    out[i] = s;
}

// ---------- CRF log-likelihood: one wave per batch row, lanes 0..8 hold alpha[c]
__global__ void k_crf(const float* __restrict__ logits, const int* __restrict__ tags,
                      const int* __restrict__ len, const float* __restrict__ trans,
                      float* __restrict__ ll) {
    int b = blockIdx.x, lane = threadIdx.x;
    int L = len[b];
    const float* lg = logits + (size_t)b * T * C;
    float tcol[C];
    if (lane < C)
        for (int k = 0; k < C; ++k) tcol[k] = trans[k * C + lane];
    float alpha = (lane < C) ? lg[lane] : -1e30f;
    for (int t = 1; t < T; ++t) {
        float av[C];
#pragma unroll
        for (int k = 0; k < C; ++k) av[k] = __shfl(alpha, k, 32);
        if (lane < C) {
            float mx = -1e30f;
#pragma unroll
            for (int k = 0; k < C; ++k) mx = fmaxf(mx, av[k] + tcol[k]);
            float s = 0.f;
#pragma unroll
            for (int k = 0; k < C; ++k) s += __expf(av[k] + tcol[k] - mx);
            float nv = mx + __logf(s) + lg[(size_t)t * C + lane];
            if (t < L) alpha = nv;
        }
    }
    float av[C];
#pragma unroll
    for (int k = 0; k < C; ++k) av[k] = __shfl(alpha, k, 32);
    float mx = -1e30f;
#pragma unroll
    for (int k = 0; k < C; ++k) mx = fmaxf(mx, av[k]);
    float s = 0.f;
#pragma unroll
    for (int k = 0; k < C; ++k) s += __expf(av[k] - mx);
    float log_norm = mx + __logf(s);

    const int* tg = tags + (size_t)b * T;
    float u = 0.f;
    for (int t = lane; t < T; t += 32)
        if (t < L) u += lg[(size_t)t * C + tg[t]];
    float bin = 0.f;
    for (int t = lane; t < T - 1; t += 32)
        if (t < L - 1) bin += trans[tg[t] * C + tg[t + 1]];
    float tot = u + bin;
#pragma unroll
    for (int off = 16; off; off >>= 1) tot += __shfl_down(tot, off, 32);
    if (lane == 0) ll[b] = tot - log_norm;
}

__global__ void k_tail(const float* __restrict__ trans, float* __restrict__ out) {
    int i = threadIdx.x;
    if (i < C * C) out[i] = trans[i];
}

extern "C" void kernel_launch(void* const* d_in, const int* in_sizes, int n_in,
                              void* d_out, int out_size, void* d_ws, size_t ws_size,
                              hipStream_t stream) {
    (void)in_sizes; (void)n_in; (void)out_size; (void)ws_size;
    const int*   inputs    = (const int*)  d_in[0];
    const int*   lengths   = (const int*)  d_in[1];
    const int*   targets   = (const int*)  d_in[2];
    const float* embedding = (const float*)d_in[3];
    const float* Wk_f = (const float*)d_in[4];
    const float* Wr_f = (const float*)d_in[5];
    const float* b_f  = (const float*)d_in[6];
    const float* Wk_b = (const float*)d_in[7];
    const float* Wr_b = (const float*)d_in[8];
    const float* b_b  = (const float*)d_in[9];
    const float* dW   = (const float*)d_in[10];
    const float* db   = (const float*)d_in[11];
    const float* trans= (const float*)d_in[12];

    unsigned char* ws = (unsigned char*)d_ws;
    size_t off = 0;
    auto alloc = [&](size_t bytes) -> void* {
        void* p = ws + off;
        off += (bytes + 255) & ~(size_t)255;
        return p;
    };
    unsigned short* embBF = (unsigned short*)alloc((size_t)BT * Dp * 2);
    unsigned short* WkTf  = (unsigned short*)alloc((size_t)NH * Dp * 2);
    unsigned short* WkTb  = (unsigned short*)alloc((size_t)NH * Dp * 2);
    unsigned short* WrTf  = (unsigned short*)alloc((size_t)NH * H * 2);
    unsigned short* WrTb  = (unsigned short*)alloc((size_t)NH * H * 2);
    float*          xzf   = (float*)alloc((size_t)BT * NH * 4);
    float*          xzb   = (float*)alloc((size_t)BT * NH * 4);
    unsigned short* hF    = (unsigned short*)alloc((size_t)BT * H * 2);
    unsigned short* hB    = (unsigned short*)alloc((size_t)BT * H * 2);

    float* logits = (float*)d_out;
    float* ll     = logits + (size_t)BT * C;
    float* transO = ll + B;

    // 1) weight conversion (transpose + pad + bf16)
    {
        int nWk = NH * Dp, nWr = NH * H;
        k_convT<<<(nWk + 255) / 256, 256, 0, stream>>>(Wk_f, WkTf, D, NH, Dp);
        k_convT<<<(nWk + 255) / 256, 256, 0, stream>>>(Wk_b, WkTb, D, NH, Dp);
        k_convT<<<(nWr + 255) / 256, 256, 0, stream>>>(Wr_f, WrTf, H, NH, H);
        k_convT<<<(nWr + 255) / 256, 256, 0, stream>>>(Wr_b, WrTb, H, NH, H);
    }
    // 2) embedding gather
    k_embed<<<(BT * Dp + 255) / 256, 256, 0, stream>>>(inputs, embedding, embBF);
    // 3) xz pre-activations, both directions (WMMA GEMM, TDM + async staging)
    {
        dim3 grid(NH / 128, BT / 128);
        k_xz<<<grid, 256, 0, stream>>>(embBF, WkTf, b_f, xzf);
        k_xz<<<grid, 256, 0, stream>>>(embBF, WkTb, b_b, xzb);
    }
    // 4) recurrence: one persistent WGP per direction
    k_lstm<<<2, 1024, LSTM_SMEM, stream>>>(xzf, xzb, WrTf, WrTb, lengths, hF, hB);
    // 5) dense projection -> logits (first part of d_out)
    k_dense<<<(BT * C + 255) / 256, 256, 0, stream>>>(hF, hB, dW, db, logits);
    // 6) CRF log-likelihood
    k_crf<<<B, 32, 0, stream>>>(logits, targets, lengths, trans, ll);
    // 7) trans passthrough output
    k_tail<<<1, 128, 0, stream>>>(trans, transO);
}
